// KroneckerAttention_44461501448613
// MI455X (gfx1250) — matrix operand order repaired and checked by reading, not compiled, not run on hardware
//
#include <hip/hip_runtime.h>

// ---------------------------------------------------------------------------
// Kronecker attention, MI455X (gfx1250, wave32).
// One wave per batch iteration, persistent blocks (grid-stride).
// All matmuls: V_WMMA_F32_16X16X32_BF16 (f32 accum). Softmax/LN in f32.
// LDS operand buffers zero-padded -> unconditional, aligned b128 gathers.
// Write-heavy intermediates (Afeat, mid) stored f32 (direct acc stores);
// f32->bf16 at gather via packed v_cvt_pk_bf16_f32. Softmax row-sums fused
// into out1 as ones-A WMMAs; 1/sum deferred into out1's column scale.
// 1/sqrt(d_k)=0.25 folded into staged Wq weights (exact in bf16).
// ---------------------------------------------------------------------------

typedef __bf16 bf16_t;
typedef __attribute__((ext_vector_type(16))) __bf16        v16bf;
typedef __attribute__((ext_vector_type(2)))  __bf16        v2bf;
typedef __attribute__((ext_vector_type(8)))  float         v8f;
typedef __attribute__((ext_vector_type(8)))  unsigned int  v8u;
typedef __attribute__((ext_vector_type(4)))  unsigned int  v4u;

#define WAVES    4
#define NTHREADS (32 * WAVES)
#define LN_EPS   1e-5f

struct WaveScratch {
  bf16_t Xbf[16 * 64];    // X row-major, rows 8..15 stay zero
  bf16_t Xtbf[64 * 32];   // X^T (64x8), cols 8..31 stay zero
  bf16_t Qch[16 * 16];    // channel Q (A-operand only, K<=16 folds)
  bf16_t Kch[16 * 32];    // channel K, cols 16..31 stay zero (B-operand)
  bf16_t Qf[64 * 16];     // feature Q (A-operand only; Wq pre-scaled 0.25)
  bf16_t Kf[64 * 32];     // feature K, cols 16..31 stay zero (B-operand)
  bf16_t Ach[16 * 16];    // channel attention, zero-padded
  bf16_t out1bf[16 * 64]; // mid @ A_feat^T (rows 8..15 naturally zero)
  float  midbf[16 * 64];  // A_ch @ X, f32 (rows 8..15 naturally zero)
  float  Afeat[64 * 64];  // exp(scores), f32; first 4KB doubles as `strip`
};                        // 36 KB

struct BlockShared {
  bf16_t Wq_ch[16 * 64];  // pre-scaled by 0.25
  bf16_t Wk_ch[16 * 64];
  bf16_t Wqf[16 * 32];    // pre-scaled by 0.25, zero-padded K 8->32
  bf16_t Wkf[16 * 32];
  bf16_t Wout[64 * 64];
  WaveScratch ws[WAVES];
};                        // ~158 KB (two blocks fit a 320KB WGP)

// --- WMMA operand lane layouts (CDNA5 ISA §7.12.2, wave32) ------
// A (16x32): lane holds row m = lane%16; b128 quads {half, 2+half}.
// B (32x16): lane holds col n = lane%16 of B = src^T; quads {half, 2+half}.
// C/D (16x16 f32): VGPR r holds (m = r + 8*half, n = lane%16).

struct q2 { v4u lo, hi; };

__device__ __forceinline__ v16bf ld_A32(const bf16_t* src, int ld, int half, int lm) {
  const v4u* p = (const v4u*)(src + lm * ld);
  q2 t;
  t.lo = p[half];
  t.hi = p[2 + half];
  return __builtin_bit_cast(v16bf, t);
}

__device__ __forceinline__ v16bf ld_A16(const bf16_t* src, int ld, int half, int lm) {
  const v4u* p = (const v4u*)(src + lm * ld);
  q2 t;
  t.lo = p[half];
  t.hi = v4u{0u, 0u, 0u, 0u};
  return __builtin_bit_cast(v16bf, t);
}

__device__ __forceinline__ v16bf ld_B32(const bf16_t* src, int ld, int half, int lm) {
  const v4u* p = (const v4u*)(src + lm * ld);
  q2 t;
  t.lo = p[half];
  t.hi = p[2 + half];
  return __builtin_bit_cast(v16bf, t);
}

// packed f32 pair -> bf16 pair (pattern-matches v_cvt_pk_bf16_f32)
__device__ __forceinline__ unsigned int pack2(float a, float b) {
  v2bf t = {(bf16_t)a, (bf16_t)b};
  return __builtin_bit_cast(unsigned int, t);
}

__device__ __forceinline__ v16bf cvt16(float4 f0, float4 f1, float4 f2, float4 f3) {
  v8u t;
  t[0] = pack2(f0.x, f0.y); t[1] = pack2(f0.z, f0.w);
  t[2] = pack2(f1.x, f1.y); t[3] = pack2(f1.z, f1.w);
  t[4] = pack2(f2.x, f2.y); t[5] = pack2(f2.z, f2.w);
  t[6] = pack2(f3.x, f3.y); t[7] = pack2(f3.z, f3.w);
  return __builtin_bit_cast(v16bf, t);
}

__device__ __forceinline__ v16bf ld_A32_f(const float* src, int ld, int half, int lm) {
  const float4* p = (const float4*)(src + lm * ld);
  return cvt16(p[2 * half], p[2 * half + 1], p[4 + 2 * half], p[5 + 2 * half]);
}

__device__ __forceinline__ v16bf ld_B32_f(const float* src, int ld, int half, int lm) {
  const float4* p = (const float4*)(src + lm * ld);
  return cvt16(p[2 * half], p[2 * half + 1], p[4 + 2 * half], p[5 + 2 * half]);
}

__device__ __forceinline__ v8f wmma_bf(const v16bf& a, const v16bf& b, const v8f& c) {
  return __builtin_amdgcn_wmma_f32_16x16x32_bf16(false, a, false, b,
                                                 (short)0, c, false, false);
}

__device__ __forceinline__ void store_D_bf(bf16_t* dst, int ld, v8f d, int half, int lm) {
#pragma unroll
  for (int r = 0; r < 8; ++r) dst[(r + 8 * half) * ld + lm] = (bf16_t)d[r];
}

__device__ __forceinline__ void store_D_f32(float* dst, int ld, v8f d, int half, int lm) {
#pragma unroll
  for (int r = 0; r < 8; ++r) dst[(r + 8 * half) * ld + lm] = d[r];
}

__global__ __launch_bounds__(NTHREADS) void kron_attn_kernel(
    const float* __restrict__ X, const float* __restrict__ Wq_ch,
    const float* __restrict__ Wk_ch, const float* __restrict__ Wq_feat,
    const float* __restrict__ Wk_feat, const float* __restrict__ W_out,
    const float* __restrict__ ln_w, const float* __restrict__ ln_b,
    float* __restrict__ out, int nbatch) {
  extern __shared__ __align__(16) char smem_raw[];
  BlockShared& sh = *reinterpret_cast<BlockShared*>(smem_raw);

  const int tid  = threadIdx.x;
  const int wave = tid >> 5;
  const int lane = tid & 31;
  const int half = lane >> 4;
  const int lm   = lane & 15;

  // -------- stage weights (bf16, padded, Wq pre-scaled) once per block -----
  for (int i = tid; i < 16 * 64; i += NTHREADS) {
    sh.Wq_ch[i] = (bf16_t)(0.25f * Wq_ch[i]);  // 1/sqrt(d_k) folded in
    sh.Wk_ch[i] = (bf16_t)Wk_ch[i];
  }
  for (int i = tid; i < 16 * 32; i += NTHREADS) {
    int n = i >> 5, k = i & 31;
    sh.Wqf[i] = (k < 8) ? (bf16_t)(0.25f * Wq_feat[n * 8 + k]) : (bf16_t)0.0f;
    sh.Wkf[i] = (k < 8) ? (bf16_t)Wk_feat[n * 8 + k] : (bf16_t)0.0f;
  }
  for (int i = tid; i < 64 * 64; i += NTHREADS) sh.Wout[i] = (bf16_t)W_out[i];

  WaveScratch& w = sh.ws[wave];
  float* strip = w.Afeat;  // lifetime-disjoint alias (channel scores / pre-LN)
  {  // zero entire wave scratch once: pads stay zero across batch iterations
    unsigned int* p = (unsigned int*)&w;
    const int n = (int)(sizeof(WaveScratch) / 4);
    for (int i = lane; i < n; i += 32) p[i] = 0u;
  }
  __syncthreads();

  const v8f Z = {0.f, 0.f, 0.f, 0.f, 0.f, 0.f, 0.f, 0.f};
  v8u ones_u;
#pragma unroll
  for (int v = 0; v < 8; ++v) ones_u[v] = 0x3F803F80u;  // bf16 1.0 pairs
  const v16bf ONES = __builtin_bit_cast(v16bf, ones_u);

  const int stride = gridDim.x * WAVES;

  for (int batch = blockIdx.x * WAVES + wave; batch < nbatch; batch += stride) {
    // ---- stage X: lane-contiguous regs (matches LN layout) + LDS copies ---
    float xr[16];
    const float* Xb = X + (size_t)batch * 512;
#pragma unroll
    for (int i = 0; i < 4; ++i) {
      float4 v = *(const float4*)(Xb + lane * 16 + i * 4);
      xr[4 * i + 0] = v.x; xr[4 * i + 1] = v.y;
      xr[4 * i + 2] = v.z; xr[4 * i + 3] = v.w;
    }
    {  // row-major bf16 copy: two b128 stores per lane
      v4u x0, x1;
#pragma unroll
      for (int i = 0; i < 4; ++i) {
        x0[i] = pack2(xr[2 * i], xr[2 * i + 1]);
        x1[i] = pack2(xr[8 + 2 * i], xr[9 + 2 * i]);
      }
      ((v4u*)w.Xbf)[lane * 2 + 0] = x0;
      ((v4u*)w.Xbf)[lane * 2 + 1] = x1;
    }
#pragma unroll
    for (int i = 0; i < 16; ++i) {  // transposed copy: Xt[d][k] = X[k][d]
      int flat = lane * 16 + i;
      w.Xtbf[(flat & 63) * 32 + (flat >> 6)] = (bf16_t)xr[i];
    }

    // ---- channel attention: Q_ch/K_ch = X @ W^T (Q scale pre-folded) ------
    {
      v8f q = Z, k = Z;
#pragma unroll
      for (int c = 0; c < 2; ++c) {
        v16bf a = ld_A32(w.Xbf + 32 * c, 64, half, lm);
        q = wmma_bf(a, ld_B32(sh.Wq_ch + 32 * c, 64, half, lm), q);
        k = wmma_bf(a, ld_B32(sh.Wk_ch + 32 * c, 64, half, lm), k);
      }
      store_D_bf(w.Qch, 16, q, half, lm);
      store_D_bf(w.Kch, 32, k, half, lm);
    }
    {  // scores_ch = Q_ch @ K_ch^T (d_k=16 zero-padded; scale pre-folded)
      v8f s = wmma_bf(ld_A16(w.Qch, 16, half, lm), ld_B32(w.Kch, 32, half, lm), Z);
      store_D_f32(strip, 16, s, half, lm);
    }
    // softmax over valid 8x8 -> A_ch (rest of 16x16 zeroed); no max needed
#pragma unroll
    for (int i = 0; i < 4; ++i) ((unsigned int*)w.Ach)[lane + 32 * i] = 0u;
    if (lane < 8) {
      float4 va = *(const float4*)(strip + lane * 16);
      float4 vb = *(const float4*)(strip + lane * 16 + 4);
      float e0 = __expf(va.x), e1 = __expf(va.y);
      float e2 = __expf(va.z), e3 = __expf(va.w);
      float e4 = __expf(vb.x), e5 = __expf(vb.y);
      float e6 = __expf(vb.z), e7 = __expf(vb.w);
      float inv = __builtin_amdgcn_rcpf(((e0 + e1) + (e2 + e3)) +
                                        ((e4 + e5) + (e6 + e7)));
      v4u p;
      p[0] = pack2(e0 * inv, e1 * inv);
      p[1] = pack2(e2 * inv, e3 * inv);
      p[2] = pack2(e4 * inv, e5 * inv);
      p[3] = pack2(e6 * inv, e7 * inv);
      ((v4u*)w.Ach)[lane * 2] = p;  // cols 0..7 of row `lane`
    }

    // ---- feature Q/K: X^T (64x8, padded) @ W_feat^T (Q scale pre-folded) --
    {
      v16bf bq = ld_B32(sh.Wqf, 32, half, lm);
      v16bf bk = ld_B32(sh.Wkf, 32, half, lm);
#pragma unroll
      for (int ti = 0; ti < 4; ++ti) {
        v16bf a = ld_A32(w.Xtbf + ti * 512, 32, half, lm);
        store_D_bf(w.Qf + ti * 256, 16, wmma_bf(a, bq, Z), half, lm);
        store_D_bf(w.Kf + ti * 512, 32, wmma_bf(a, bk, Z), half, lm);
      }
    }

    // ---- feature scores -> exp, stored f32 (no max: scores O(10)) --------
    {
      v16bf kb0 = ld_B32(w.Kf + 0 * 512, 32, half, lm);
      v16bf kb1 = ld_B32(w.Kf + 1 * 512, 32, half, lm);
      v16bf kb2 = ld_B32(w.Kf + 2 * 512, 32, half, lm);
      v16bf kb3 = ld_B32(w.Kf + 3 * 512, 32, half, lm);
#pragma unroll
      for (int ti = 0; ti < 4; ++ti) {
        v16bf a = ld_A16(w.Qf + ti * 256, 16, half, lm);
        v8f s0 = wmma_bf(a, kb0, Z);
        v8f s1 = wmma_bf(a, kb1, Z);
        v8f s2 = wmma_bf(a, kb2, Z);
        v8f s3 = wmma_bf(a, kb3, Z);
#pragma unroll
        for (int r = 0; r < 8; ++r) {
          s0[r] = __expf(s0[r]);
          s1[r] = __expf(s1[r]);
          s2[r] = __expf(s2[r]);
          s3[r] = __expf(s3[r]);
        }
        float* ab = w.Afeat + ti * 1024;
        store_D_f32(ab + 0,  64, s0, half, lm);
        store_D_f32(ab + 16, 64, s1, half, lm);
        store_D_f32(ab + 32, 64, s2, half, lm);
        store_D_f32(ab + 48, 64, s3, half, lm);
      }
    }

    // ---- mid = A_ch @ X  (B from transposed copy), f32 store --------------
    {
      v16bf a = ld_A16(w.Ach, 16, half, lm);
#pragma unroll
      for (int j = 0; j < 4; ++j) {
        v8f s = wmma_bf(a, ld_B32(w.Xtbf + j * 512, 32, half, lm), Z);
        store_D_f32(w.midbf + j * 16, 64, s, half, lm);
      }
    }

    // ---- out1 = mid @ A_feat^T with fused row-sum (ones-A WMMA) -----------
    {
      v16bf a0 = ld_A32_f(w.midbf, 64, half, lm);
      v16bf a1 = ld_A32_f(w.midbf + 32, 64, half, lm);
#pragma unroll
      for (int j = 0; j < 4; ++j) {
        v16bf b0 = ld_B32_f(w.Afeat + j * 1024, 64, half, lm);
        v16bf b1 = ld_B32_f(w.Afeat + j * 1024 + 32, 64, half, lm);
        v8f s = wmma_bf(a0, b0, Z);
        s = wmma_bf(a1, b1, s);
        v8f rs = wmma_bf(ONES, b0, Z);
        rs = wmma_bf(ONES, b1, rs);
        float sc = __builtin_amdgcn_rcpf(rs[0]);
#pragma unroll
        for (int r = 0; r < 8; ++r) s[r] *= sc;
        store_D_bf(w.out1bf + j * 16, 64, s, half, lm);
      }
    }

    // ---- out2 = out1 @ W_out^T (Afeat dead -> strip alias safe) -----------
    {
      v16bf a0 = ld_A32(w.out1bf, 64, half, lm);
      v16bf a1 = ld_A32(w.out1bf + 32, 64, half, lm);
#pragma unroll
      for (int j = 0; j < 4; ++j) {
        v8f s = wmma_bf(a0, ld_B32(sh.Wout + j * 1024, 64, half, lm), Z);
        s = wmma_bf(a1, ld_B32(sh.Wout + j * 1024 + 32, 64, half, lm), s);
        store_D_f32(strip + j * 16, 64, s, half, lm);
      }
    }

    // ---- residual (regs) + LayerNorm over 512 + affine + store ------------
    {
      float y[16];
      float s1 = 0.f, s2 = 0.f;
#pragma unroll
      for (int i = 0; i < 4; ++i) {
        float4 v4 = *(const float4*)(strip + lane * 16 + i * 4);
        y[4 * i + 0] = v4.x + xr[4 * i + 0];
        y[4 * i + 1] = v4.y + xr[4 * i + 1];
        y[4 * i + 2] = v4.z + xr[4 * i + 2];
        y[4 * i + 3] = v4.w + xr[4 * i + 3];
      }
#pragma unroll
      for (int i = 0; i < 16; ++i) {
        s1 += y[i];
        s2 += y[i] * y[i];
      }
#pragma unroll
      for (int off = 16; off > 0; off >>= 1) {
        s1 += __shfl_xor(s1, off, 32);
        s2 += __shfl_xor(s2, off, 32);
      }
      float mu   = s1 * (1.f / 512.f);
      float var  = s2 * (1.f / 512.f) - mu * mu;
      float rstd = rsqrtf(var + LN_EPS);
      float* ob = out + (size_t)batch * 512 + lane * 16;
      const float* lw = ln_w + lane * 16;
      const float* lb = ln_b + lane * 16;
#pragma unroll
      for (int i = 0; i < 16; i += 4) {
        float4 gw = *(const float4*)(lw + i);
        float4 gb = *(const float4*)(lb + i);
        float4 o;
        o.x = (y[i + 0] - mu) * rstd * gw.x + gb.x;
        o.y = (y[i + 1] - mu) * rstd * gw.y + gb.y;
        o.z = (y[i + 2] - mu) * rstd * gw.z + gb.z;
        o.w = (y[i + 3] - mu) * rstd * gw.w + gb.w;
        *(float4*)(ob + i) = o;
      }
    }
  }
}

extern "C" void kernel_launch(void* const* d_in, const int* in_sizes, int n_in,
                              void* d_out, int out_size, void* d_ws, size_t ws_size,
                              hipStream_t stream) {
  const float* X       = (const float*)d_in[0];
  const float* Wq_ch   = (const float*)d_in[1];
  const float* Wk_ch   = (const float*)d_in[2];
  const float* Wq_feat = (const float*)d_in[3];
  const float* Wk_feat = (const float*)d_in[4];
  const float* W_out   = (const float*)d_in[5];
  const float* ln_w    = (const float*)d_in[6];
  const float* ln_b    = (const float*)d_in[7];
  float* out = (float*)d_out;

  int nbatch = in_sizes[0] / 512;  // B*K*D / (8*64)
  int slots  = (nbatch + WAVES - 1) / WAVES;
  int grid   = slots < 2048 ? slots : 2048;  // persistent blocks, grid-stride
  size_t shbytes = sizeof(BlockShared);
  kron_attn_kernel<<<grid, NTHREADS, shbytes, stream>>>(
      X, Wq_ch, Wk_ch, Wq_feat, Wk_feat, W_out, ln_w, ln_b, out, nbatch);
}
